// RecurDropLayerNormLSTM_82480551952726
// MI455X (gfx1250) — matrix operand order repaired and checked by reading, not compiled, not run on hardware
//
#include <hip/hip_runtime.h>

// ---------------------------------------------------------------------------
// 2-layer LayerNorm-LSTM, persistent-kernel design for MI455X (gfx1250).
// GEMMs run on v_wmma_f32_16x16x32_bf16 (fp32 accumulate). Weights/x/h are
// pre-packed into the CDNA5 wave32 WMMA fragment layouts. Each WG's weight
// slice (224 KB) is staged once into the 320 KB WGP LDS, so the per-step hot
// loop is ds_load_b128 (B fragments) + global b128 (A fragments) + WMMA.
// Cross-step recurrence handled by a device-scope generation barrier
// (32 co-resident workgroups, L2-backed atomics).
// ---------------------------------------------------------------------------

constexpr int kSeq = 1024;
constexpr int kB   = 64;    // batch
constexpr int kIn  = 256;   // input features
constexpr int kH   = 512;   // hidden
constexpr int kNWG = 32;    // persistent workgroups (each owns 16 hidden units)

typedef __attribute__((ext_vector_type(16))) __bf16 v16bf;
typedef __attribute__((ext_vector_type(8)))  float  v8f;

// Workspace layout (bytes)
constexpr size_t BAR_OFF = 0;                       // 2 words barrier (padded 256)
constexpr size_t HAP_OFF = 256;                     // packed h layer0: 4*16*512 bf16
constexpr size_t HBP_OFF = HAP_OFF + 65536;         // packed h layer1
constexpr size_t RED_OFF = HBP_OFF + 65536;         // LN partials: 32*64*2 f32
constexpr size_t XP_OFF  = RED_OFF + 16384;         // packed x: 1024*4*8*512 bf16
constexpr size_t W0P_OFF = XP_OFF + (size_t)kSeq * 4 * 8 * 512 * 2;   // +32 MB
constexpr size_t W1P_OFF = W0P_OFF + (size_t)128 * 24 * 512 * 2;      // +3 MB
// total ~41 MB

// Dynamic LDS layout (bytes) -- 242.5 KB of the 320 KB WGP LDS
constexpr int W0L_BYTES = 4 * 24 * 1024;            // layer0 weight slice  (96 KB)
constexpr int W1L_BYTES = 4 * 32 * 1024;            // layer1 weight slice (128 KB)
constexpr int GL_OFF    = W0L_BYTES + W1L_BYTES;    // gates [4][64][16] f32 (16 KB)
constexpr int PS_OFF    = GL_OFF + 16384;
constexpr int PS2_OFF   = PS_OFF + 1024;
constexpr int MV_OFF    = PS2_OFF + 1024;
constexpr int SMEM_TOTAL = MV_OFF + 512;            // 248320 B

// A-fragment K mapping for 16-bit A 16x32 (ISA 7.12.2):
// lanes 0-15: VGPR0-3 = K0..7, VGPR4-7 = K16..23; lanes 16-31: K8..15, K24..31
__device__ __host__ __forceinline__ int kmapA(int lane, int j) {
  return ((j < 8) ? j : j + 8) + ((lane < 16) ? 0 : 8);
}
// B-fragment K mapping for 16-bit B 32x16 (same K order as A):
// lanes 0-15: K0..7, K16..23; lanes 16-31: K8..15, K24..31
__device__ __host__ __forceinline__ int kmapB(int lane, int j) {
  return ((j < 8) ? j : j + 8) + ((lane < 16) ? 0 : 8);
}

__device__ __forceinline__ float sigmoidf(float z) {
  return 1.0f / (1.0f + __expf(-z));
}

// ----------------------------- prep kernels --------------------------------

__global__ void zero_bar_kernel(unsigned* p) {
  if (threadIdx.x < 64) p[threadIdx.x] = 0u;
}

// pack x[t][b][k] (fp32) -> A fragments (bf16), blocks of 512 elems per
// (t, mtile, ktile); lane stride 16 elems (32 B) -> two b128 loads per frag.
__global__ void pack_x_kernel(const float* __restrict__ x, __bf16* __restrict__ xp) {
  const int total = kSeq * 4 * 8 * 512;
  for (int idx = blockIdx.x * blockDim.x + threadIdx.x; idx < total;
       idx += gridDim.x * blockDim.x) {
    int j = idx & 15, lane = (idx >> 4) & 31, kt = (idx >> 9) & 7,
        mt = (idx >> 12) & 3, t = idx >> 14;
    int b = mt * 16 + (lane & 15);
    int k = kt * 32 + kmapA(lane, j);
    xp[idx] = (__bf16)x[((long)t * kB + b) * kIn + k];
  }
}

// pack h[b][k] (fp32, 64x512) -> A fragments, 4 mtiles x 16 ktiles
__global__ void pack_h_kernel(const float* __restrict__ h, __bf16* __restrict__ hp) {
  const int total = 4 * 16 * 512;
  for (int idx = blockIdx.x * blockDim.x + threadIdx.x; idx < total;
       idx += gridDim.x * blockDim.x) {
    int j = idx & 15, lane = (idx >> 4) & 31, kt = (idx >> 9) & 15, mt = idx >> 13;
    int b = mt * 16 + (lane & 15);
    int k = kt * 32 + kmapA(lane, j);
    hp[idx] = (__bf16)h[(long)b * kH + k];
  }
}

// pack W[n][k] (torch Linear layout [4H, Kdim]) -> B fragments (B = W^T tile)
__global__ void pack_w_kernel(const float* __restrict__ W, __bf16* __restrict__ wp,
                              int Kdim, int KT) {
  const int total = 128 * KT * 512;
  for (int idx = blockIdx.x * blockDim.x + threadIdx.x; idx < total;
       idx += gridDim.x * blockDim.x) {
    int j = idx & 15, lane = (idx >> 4) & 31;
    int kt = (idx >> 9) % KT;
    int nt = idx / (512 * KT);
    int n = nt * 16 + (lane & 15);
    int k = kt * 32 + kmapB(lane, j);
    wp[idx] = (__bf16)W[(long)n * Kdim + k];
  }
}

// --------------------------- device barrier --------------------------------

__device__ void grid_barrier(unsigned* bar, unsigned nwg) {
  __threadfence();          // publish this thread's global stores
  __syncthreads();
  if (threadIdx.x == 0) {
    unsigned gen = __hip_atomic_load(bar + 1, __ATOMIC_RELAXED, __HIP_MEMORY_SCOPE_AGENT);
    unsigned arr = __hip_atomic_fetch_add(bar, 1u, __ATOMIC_ACQ_REL, __HIP_MEMORY_SCOPE_AGENT);
    if (arr == nwg - 1u) {
      __hip_atomic_store(bar, 0u, __ATOMIC_RELAXED, __HIP_MEMORY_SCOPE_AGENT);
      __hip_atomic_fetch_add(bar + 1, 1u, __ATOMIC_RELEASE, __HIP_MEMORY_SCOPE_AGENT);
    } else {
      while (__hip_atomic_load(bar + 1, __ATOMIC_ACQUIRE, __HIP_MEMORY_SCOPE_AGENT) == gen) {
        __builtin_amdgcn_s_sleep(2);
      }
    }
  }
  __syncthreads();
}

// ------------------------------ LSTM cell ----------------------------------
// One cell step for this WG's 16 hidden units, all 64 batch rows.
// A = [a0 (A0K ktiles) | a1 (KT-A0K ktiles, stride 16 blocks per mtile)]
// B fragments come from the LDS-resident weight slice (wlds).
template <int KT, int A0K>
__device__ __forceinline__ void run_cell(
    const __bf16* __restrict__ a0, const __bf16* __restrict__ a1,
    const char* wlds, __bf16* __restrict__ hdst,
    const float* bv, const float* gm, const float* bt,
    float* creg, float* hreg,
    float* redbuf, unsigned* bar,
    float* gl, float* ps, float* ps2, float* mv,
    int wg, int tid) {
  const int lane  = tid & 31;
  const int wave  = tid >> 5;
  const int mtile = wave & 3;          // batch tile 0..3
  const int gpair = (wave >> 2) << 1;  // gates {gpair, gpair+1}

  // ---- GEMM: 2 output tiles per wave, shared A fragment ----
  v8f acc0 = {};
  v8f acc1 = {};
  const char* wl0 = wlds + (size_t)(gpair * KT) * 1024 + (size_t)lane * 32;
  const char* wl1 = wlds + (size_t)((gpair + 1) * KT) * 1024 + (size_t)lane * 32;
#pragma unroll 4
  for (int kt = 0; kt < A0K; ++kt) {
    v16bf av  = *(const v16bf*)(a0 + (mtile * A0K + kt) * 512 + lane * 16);
    v16bf bv0 = *(const v16bf*)(wl0 + kt * 1024);
    v16bf bv1 = *(const v16bf*)(wl1 + kt * 1024);
    acc0 = __builtin_amdgcn_wmma_f32_16x16x32_bf16(false, av, false, bv0, (short)0, acc0, false, false);
    acc1 = __builtin_amdgcn_wmma_f32_16x16x32_bf16(false, av, false, bv1, (short)0, acc1, false, false);
  }
#pragma unroll 4
  for (int kt = A0K; kt < KT; ++kt) {
    v16bf av  = *(const v16bf*)(a1 + (mtile * 16 + (kt - A0K)) * 512 + lane * 16);
    v16bf bv0 = *(const v16bf*)(wl0 + kt * 1024);
    v16bf bv1 = *(const v16bf*)(wl1 + kt * 1024);
    acc0 = __builtin_amdgcn_wmma_f32_16x16x32_bf16(false, av, false, bv0, (short)0, acc0, false, false);
    acc1 = __builtin_amdgcn_wmma_f32_16x16x32_bf16(false, av, false, bv1, (short)0, acc1, false, false);
  }
  // C layout: lanes 0-15 N=lane M=r, lanes 16-31 N=lane-16 M=r+8
  const int mrow = (lane >> 4) << 3;
  const int ncol = lane & 15;
#pragma unroll
  for (int r = 0; r < 8; ++r) {
    gl[((gpair * 64) + mtile * 16 + mrow + r) * 16 + ncol]       = acc0[r];
    gl[(((gpair + 1) * 64) + mtile * 16 + mrow + r) * 16 + ncol] = acc1[r];
  }
  __syncthreads();

  // ---- elementwise: thread handles (b = tid%64, 4 hidden of this slice) ----
  const int eb  = tid & 63;
  const int enb = tid >> 6;
  float oreg[4], cnew[4];
  float lsum = 0.f, lsq = 0.f;
#pragma unroll
  for (int q = 0; q < 4; ++q) {
    int n = enb * 4 + q;
    float zi = gl[(0 * 64 + eb) * 16 + n] + bv[0 * 4 + q];
    float zf = gl[(1 * 64 + eb) * 16 + n] + bv[1 * 4 + q];
    float zo = gl[(2 * 64 + eb) * 16 + n] + bv[2 * 4 + q];
    float zc = gl[(3 * 64 + eb) * 16 + n] + bv[3 * 4 + q];
    float c  = sigmoidf(zi) * tanhf(zc) + sigmoidf(zf) * creg[q];
    creg[q] = c;
    cnew[q] = c;
    oreg[q] = sigmoidf(zo);
    lsum += c;
    lsq  += c * c;
  }
  ps[tid]  = lsum;
  ps2[tid] = lsq;
  __syncthreads();
  if (tid < 64) {  // per-batch partial over this WG's 16 hidden
    float s  = ps[tid] + ps[tid + 64] + ps[tid + 128] + ps[tid + 192];
    float s2 = ps2[tid] + ps2[tid + 64] + ps2[tid + 128] + ps2[tid + 192];
    redbuf[(wg * 64 + tid) * 2 + 0] = s;
    redbuf[(wg * 64 + tid) * 2 + 1] = s2;
  }
  grid_barrier(bar, kNWG);  // all partials visible; all GEMM reads of old h done

  if (tid < 64) {  // full LayerNorm statistics over H=512
    float s = 0.f, s2 = 0.f;
    for (int w = 0; w < kNWG; ++w) {
      s  += redbuf[(w * 64 + tid) * 2 + 0];
      s2 += redbuf[(w * 64 + tid) * 2 + 1];
    }
    float mu  = s * (1.0f / (float)kH);
    float var = s2 * (1.0f / (float)kH) - mu * mu;
    mv[tid * 2 + 0] = mu;
    mv[tid * 2 + 1] = rsqrtf(var + 1e-5f);
  }
  __syncthreads();
  const float mu   = mv[eb * 2 + 0];
  const float rstd = mv[eb * 2 + 1];
#pragma unroll
  for (int q = 0; q < 4; ++q) {
    int n = enb * 4 + q;
    float hn = oreg[q] * tanhf((cnew[q] - mu) * rstd * gm[q] + bt[q]);
    hreg[q] = hn;
    // scatter h into packed A-fragment layout (inverse of kmapA)
    int kk     = ((wg & 1) << 4) + n;                       // K within ktile wg/2
    int lane_p = (eb & 15) + (((kk & 15) < 8) ? 0 : 16);
    int jj     = (kk & 7) + ((kk & 16) ? 8 : 0);
    hdst[((eb >> 4) * 16 + (wg >> 1)) * 512 + lane_p * 16 + jj] = (__bf16)hn;
  }
  grid_barrier(bar, kNWG);  // new packed h published for next GEMM
}

// --------------------------- persistent kernel -----------------------------

__global__ __launch_bounds__(256, 1) void lstm_persistent(
    const float* __restrict__ c0in,
    const float* __restrict__ bias0, const float* __restrict__ gamma0, const float* __restrict__ beta0,
    const float* __restrict__ bias1, const float* __restrict__ gamma1, const float* __restrict__ beta1,
    const __bf16* __restrict__ xp, const __bf16* __restrict__ w0p, const __bf16* __restrict__ w1p,
    __bf16* hap, __bf16* hbp, float* redbuf, unsigned* bar, float* __restrict__ out) {
  extern __shared__ __align__(16) char smem[];
  char*  w0l = smem;                         // 96 KB layer0 weight slice
  char*  w1l = smem + W0L_BYTES;             // 128 KB layer1 weight slice
  float* gl  = (float*)(smem + GL_OFF);      // gates [4][64][16]
  float* ps  = (float*)(smem + PS_OFF);
  float* ps2 = (float*)(smem + PS2_OFF);
  float* mv  = (float*)(smem + MV_OFF);      // per-batch (mu, rstd)

  const int tid = threadIdx.x;
  const int wg  = blockIdx.x;
  const int eb  = tid & 63;
  const int enb = tid >> 6;
  const int nb  = wg * 16 + enb * 4;  // first global hidden idx of this thread

  // ---- one-time: stage this WG's weight slices into LDS (amortized 1024x) ----
  // layer0: gate-tile g -> global ntile (g*32+wg), 24 KB contiguous per tile
  for (int i = tid; i < (W0L_BYTES >> 4); i += 256) {
    int g = i / 1536;                  // 1536 = 24576/16 chunks per gate tile
    int r = i - g * 1536;
    *(uint4*)(w0l + (size_t)i * 16) =
        *(const uint4*)((const char*)w0p + (size_t)(g * 32 + wg) * 24576 + (size_t)r * 16);
  }
  // layer1: 32 KB contiguous per gate tile
  for (int i = tid; i < (W1L_BYTES >> 4); i += 256) {
    int g = i >> 11;                   // 2048 chunks per gate tile
    int r = i & 2047;
    *(uint4*)(w1l + (size_t)i * 16) =
        *(const uint4*)((const char*)w1p + (size_t)(g * 32 + wg) * 32768 + (size_t)r * 16);
  }

  float bv0[16], bv1[16], gm0[4], bt0[4], gm1[4], bt1[4];
#pragma unroll
  for (int g = 0; g < 4; ++g)
#pragma unroll
    for (int q = 0; q < 4; ++q) {
      bv0[g * 4 + q] = bias0[g * kH + nb + q];
      bv1[g * 4 + q] = bias1[g * kH + nb + q];
    }
#pragma unroll
  for (int q = 0; q < 4; ++q) {
    gm0[q] = gamma0[nb + q]; bt0[q] = beta0[nb + q];
    gm1[q] = gamma1[nb + q]; bt1[q] = beta1[nb + q];
  }

  float c0r[4], c1r[4], h0r[4], h1r[4];
#pragma unroll
  for (int q = 0; q < 4; ++q) {
    c0r[q] = c0in[(long)eb * kH + nb + q];
    c1r[q] = c0in[(long)(kB + eb) * kH + nb + q];
  }
  __syncthreads();  // weight slices resident before first GEMM

  for (int t = 0; t < kSeq; ++t) {
    // layer 0: A = [x_t (8 ktiles) | h_a (16 ktiles)], K=768
    run_cell<24, 8>(xp + (long)t * (4 * 8 * 512), hap, w0l, hap,
                    bv0, gm0, bt0, c0r, h0r, redbuf, bar, gl, ps, ps2, mv, wg, tid);
    // layer 1: A = [h_a_new (16 ktiles) | h_b (16 ktiles)], K=1024
    run_cell<32, 16>(hap, hbp, w1l, hbp,
                     bv1, gm1, bt1, c1r, h1r, redbuf, bar, gl, ps, ps2, mv, wg, tid);

#pragma unroll
    for (int q = 0; q < 4; ++q)
      out[((long)t * kB + eb) * kH + nb + q] = h1r[q];

    if (t == kSeq - 1) {
      const long HF = (long)kSeq * kB * kH;
#pragma unroll
      for (int q = 0; q < 4; ++q) {
        out[HF + (long)eb * kH + nb + q]            = h0r[q];  // h_final[0]
        out[HF + (long)(kB + eb) * kH + nb + q]     = h1r[q];  // h_final[1]
        out[HF + (long)(2 * kB + eb) * kH + nb + q] = c0r[q];  // c_final[0]
        out[HF + (long)(3 * kB + eb) * kH + nb + q] = c1r[q];  // c_final[1]
      }
    }
  }
}

// ------------------------------- launcher ----------------------------------

extern "C" void kernel_launch(void* const* d_in, const int* in_sizes, int n_in,
                              void* d_out, int out_size, void* d_ws, size_t ws_size,
                              hipStream_t stream) {
  const float* x   = (const float*)d_in[0];
  const float* h0  = (const float*)d_in[1];
  const float* c0  = (const float*)d_in[2];
  const float* W0  = (const float*)d_in[3];
  const float* b0  = (const float*)d_in[4];
  const float* g0  = (const float*)d_in[5];
  const float* be0 = (const float*)d_in[6];
  const float* W1  = (const float*)d_in[7];
  const float* b1  = (const float*)d_in[8];
  const float* g1  = (const float*)d_in[9];
  const float* be1 = (const float*)d_in[10];
  float* out = (float*)d_out;
  char*  ws  = (char*)d_ws;

  unsigned* bar    = (unsigned*)(ws + BAR_OFF);
  __bf16*   hap    = (__bf16*)(ws + HAP_OFF);
  __bf16*   hbp    = (__bf16*)(ws + HBP_OFF);
  float*    redbuf = (float*)(ws + RED_OFF);
  __bf16*   xp     = (__bf16*)(ws + XP_OFF);
  __bf16*   w0p    = (__bf16*)(ws + W0P_OFF);
  __bf16*   w1p    = (__bf16*)(ws + W1P_OFF);

  // allow >64 KB dynamic LDS (CDNA5 WGP has 320 KB); host-side, capture-safe
  (void)hipFuncSetAttribute(reinterpret_cast<const void*>(lstm_persistent),
                            hipFuncAttributeMaxDynamicSharedMemorySize, SMEM_TOTAL);

  // ws is poisoned between correctness call and timed replays: re-init barrier
  zero_bar_kernel<<<1, 64, 0, stream>>>(bar);
  // pack initial hidden state and all static operands into WMMA fragment layout
  pack_h_kernel<<<32, 256, 0, stream>>>(h0, hap);
  pack_h_kernel<<<32, 256, 0, stream>>>(h0 + (long)kB * kH, hbp);
  pack_x_kernel<<<2048, 256, 0, stream>>>(x, xp);
  pack_w_kernel<<<1024, 256, 0, stream>>>(W0, w0p, kIn + kH, 24);
  pack_w_kernel<<<1024, 256, 0, stream>>>(W1, w1p, kH + kH, 32);

  // persistent recurrent kernel: 32 WGs, 8 waves each, 1024 steps
  lstm_persistent<<<kNWG, 256, SMEM_TOTAL, stream>>>(c0, b0, g0, be0, b1, g1, be1,
                                                     xp, w0p, w1p, hap, hbp, redbuf, bar, out);
}
